// GCN_12979391169341
// MI455X (gfx1250) — matrix operand order repaired and checked by reading, not compiled, not run on hardware
//
#include <hip/hip_runtime.h>
#include <hip/hip_bf16.h>

// GCN graph-conv for MI455X (gfx1250, wave32, WMMA).
// Pipeline:
//   1) bf16 conversions: A1 -> A1h, conv_E -> ET (E-major), x -> XT ([512,2048])
//   2) CEL = A1 @ conv_E           (v_wmma_f32_16x16x32_bf16)
//   3) S   = softmax(relu(CEL @ conv_E^T))  (VALU + LDS + shuffles), stored bf16
//   4) G1  = S @ X ; G2 = 2*S@G1 - X   (Chebyshev identity; avoids N^3 S@S)
//      -> 16x64 register tile per wave, B panel staged in LDS via
//         global_load_async_to_lds_b128 (ASYNCcnt) when available
//   5) out[b,n,:] = [x|G1|G2][b,n,:96] @ W_n[96,64] + bias  (v_wmma_f32_16x16x4_f32)

#define N_NODES 2048
#define BATCH   16
#define C_IN    32
#define C_OUT   64
#define E_DIM   16
#define BC      512   // BATCH * C_IN
#define KW      96    // K * C_IN

#ifndef __has_builtin
#define __has_builtin(x) 0
#endif
#if __has_builtin(__builtin_amdgcn_global_load_async_to_lds_b128) && \
    __has_builtin(__builtin_amdgcn_s_wait_asynccnt)
#define USE_ASYNC_LDS 1
#else
#define USE_ASYNC_LDS 0
#endif

typedef __attribute__((ext_vector_type(16))) __bf16 bf16x16;
typedef __attribute__((ext_vector_type(8)))  float  f32x8;
typedef __attribute__((ext_vector_type(2)))  float  f32x2;
typedef __attribute__((ext_vector_type(4)))  unsigned int u32x4;
typedef __attribute__((ext_vector_type(4)))  int          v4i_t;

union BfFrag { u32x4 q[2]; bf16x16 v; };

__device__ inline unsigned short f2bf(float f) {
  unsigned int u = __float_as_uint(f);
  u += 0x7fffu + ((u >> 16) & 1u);     // round to nearest even
  return (unsigned short)(u >> 16);
}

// ---------------- conversion kernels ----------------
__global__ void k_cvt_bf16(const float* __restrict__ src,
                           unsigned short* __restrict__ dst, int n) {
  int i = blockIdx.x * blockDim.x + threadIdx.x;
  if (i < n) dst[i] = f2bf(src[i]);
}

// ET[e][m] = conv_E[m][e]   (E-major so B-frags are K-contiguous)
__global__ void k_build_ET(const float* __restrict__ convE,
                           unsigned short* __restrict__ ET) {
  int i = blockIdx.x * blockDim.x + threadIdx.x;  // E_DIM * N_NODES
  if (i >= E_DIM * N_NODES) return;
  int e = i >> 11, m = i & (N_NODES - 1);
  ET[i] = f2bf(convE[m * E_DIM + e]);
}

// XT[j][n] = x[b][n][c], j = b*32+c  (column-major view of X for B operand)
__global__ void k_build_XT(const float* __restrict__ x,
                           unsigned short* __restrict__ XT) {
  int i = blockIdx.x * blockDim.x + threadIdx.x;  // BC * N_NODES
  if (i >= BC * N_NODES) return;
  int j = i >> 11, n = i & (N_NODES - 1);
  XT[i] = f2bf(x[((j >> 5) * N_NODES + n) * C_IN + (j & 31)]);
}

// ---------------- stage 2: CEL = A1 @ conv_E (bf16 WMMA) ----------------
__global__ void k_gemm_cel(const unsigned short* __restrict__ A1h,
                           const unsigned short* __restrict__ ET,
                           float* __restrict__ CEL) {
  const int lane = threadIdx.x & 31;
  const int wave = threadIdx.x >> 5;
  const int hsel = lane >> 4;       // lane half (0: lanes 0-15, 1: 16-31)
  const int r    = lane & 15;
  const int m0   = (blockIdx.x * 8 + wave) * 16;

  const unsigned short* aRow = A1h + (size_t)(m0 + r) * N_NODES;
  const unsigned short* bRow = ET  + (size_t)r * N_NODES;   // r = E column

  f32x8 acc = {0.f, 0.f, 0.f, 0.f, 0.f, 0.f, 0.f, 0.f};
  for (int k0 = 0; k0 < N_NODES; k0 += 32) {
    BfFrag fa, fb;
    const unsigned short* pa = aRow + k0 + hsel * 8;
    fa.q[0] = *(const u32x4*)(pa);          // K = k0 + h*8 .. +7
    fa.q[1] = *(const u32x4*)(pa + 16);     // K = k0 + 16 + h*8 .. +7
    const unsigned short* pb = bRow + k0 + hsel * 8;
    fb.q[0] = *(const u32x4*)(pb);          // K = k0 + h*8 .. +7
    fb.q[1] = *(const u32x4*)(pb + 16);     // K = k0 + 16 + h*8 .. +7
    acc = __builtin_amdgcn_wmma_f32_16x16x32_bf16(false, fa.v, false, fb.v,
                                                  (short)0, acc, false, false);
  }
#pragma unroll
  for (int rr = 0; rr < 8; ++rr) {
    int m = m0 + rr + hsel * 8;             // C/D layout: lanes16-31 -> M+8
    CEL[m * E_DIM + r] = acc[rr];
  }
}

// ---------------- stage 3: row softmax(relu(CEL @ conv_E^T)) -> bf16 S ----
__global__ void k_softmax(const float* __restrict__ CEL,
                          const float* __restrict__ convE,
                          unsigned short* __restrict__ S) {
  __shared__ float sc[N_NODES];
  __shared__ float red[8];
  const int n = blockIdx.x;
  const int tid = threadIdx.x;

  float cl[E_DIM];
#pragma unroll
  for (int e = 0; e < E_DIM; ++e) cl[e] = CEL[n * E_DIM + e];

  float mx = 0.0f;                               // relu => scores >= 0
  for (int m = tid; m < N_NODES; m += 256) {
    const float* ce = convE + m * E_DIM;
    float s = 0.f;
#pragma unroll
    for (int e = 0; e < E_DIM; ++e) s += cl[e] * ce[e];
    s = fmaxf(s, 0.f);
    sc[m] = s;
    mx = fmaxf(mx, s);
  }
#pragma unroll
  for (int off = 16; off > 0; off >>= 1) mx = fmaxf(mx, __shfl_xor(mx, off, 32));
  if ((tid & 31) == 0) red[tid >> 5] = mx;
  __syncthreads();
  if (tid == 0) {
    float m2 = red[0];
    for (int i = 1; i < 8; ++i) m2 = fmaxf(m2, red[i]);
    red[0] = m2;
  }
  __syncthreads();
  mx = red[0];
  __syncthreads();

  float sum = 0.f;
  for (int m = tid; m < N_NODES; m += 256) {
    float e = __expf(sc[m] - mx);
    sc[m] = e;
    sum += e;
  }
#pragma unroll
  for (int off = 16; off > 0; off >>= 1) sum += __shfl_xor(sum, off, 32);
  if ((tid & 31) == 0) red[tid >> 5] = sum;
  __syncthreads();
  if (tid == 0) {
    float s2 = 0.f;
    for (int i = 0; i < 8; ++i) s2 += red[i];
    red[0] = s2;
  }
  __syncthreads();
  const float inv = 1.0f / red[0];
  for (int m = tid; m < N_NODES; m += 256)
    S[(size_t)n * N_NODES + m] = f2bf(sc[m] * inv);
}

// ---------------- stage 4: G = S @ B  (bf16 WMMA), two epilogue modes -----
// Block: 256 threads = 8 waves. Block tile: M=128 (16 per wave), N=64.
// B panel [64 rows x 128 K] staged in LDS (async when available), shared by
// all 8 waves; A-frags stream straight from global (per-wave rows).
// mode 1: Gout = S@X (f32), GoutT = bf16 transpose (feeds pass 2 B operand)
// mode 2: Gout = 2*(S@G1) - x   (Chebyshev T2 applied to x)
#define KPAN 128
__global__ void k_gemm_sx(const unsigned short* __restrict__ S,
                          const unsigned short* __restrict__ Bm,
                          float* __restrict__ Gout,
                          unsigned short* __restrict__ GoutT,
                          const float* __restrict__ xin,
                          int mode) {
  __shared__ unsigned short Bpan[64 * KPAN];    // 16 KB
  const int tid  = threadIdx.x;
  const int lane = tid & 31;
  const int wave = tid >> 5;
  const int hsel = lane >> 4;
  const int r    = lane & 15;
  const int m0   = (blockIdx.y * 8 + wave) * 16;
  const int j0   = blockIdx.x * 64;

  const unsigned short* aRow = S + (size_t)(m0 + r) * N_NODES;

  f32x8 acc[4];
#pragma unroll
  for (int t = 0; t < 4; ++t)
    acc[t] = (f32x8){0.f, 0.f, 0.f, 0.f, 0.f, 0.f, 0.f, 0.f};

  for (int k0 = 0; k0 < N_NODES; k0 += KPAN) {
    // ---- fill B panel: 64 rows x 128 K = 1024 x 16B chunks, 4 per thread
#pragma unroll
    for (int c = 0; c < 4; ++c) {
      const int cid = tid + 256 * c;
      const int row = cid >> 4;            // 0..63
      const int kc  = cid & 15;            // 16B chunk within row
      const unsigned short* gsrc =
          Bm + (size_t)(j0 + row) * N_NODES + k0 + kc * 8;
      unsigned short* ldst = &Bpan[row * KPAN + kc * 8];
#if USE_ASYNC_LDS
      // args follow the global_load_lds convention: (src_global, dst_lds, ...)
      __builtin_amdgcn_global_load_async_to_lds_b128(
          (v4i_t*)(size_t)gsrc, (v4i_t*)ldst, 0, 0);
#else
      *(u32x4*)ldst = *(const u32x4*)gsrc;
#endif
    }
#if USE_ASYNC_LDS
    __builtin_amdgcn_s_wait_asynccnt(0);
#endif
    __syncthreads();

    __builtin_prefetch(aRow + k0 + KPAN, 0, 1);   // next A panel
    // ---- 4 wmma-K steps over the panel
#pragma unroll
    for (int kk = 0; kk < KPAN / 32; ++kk) {
      BfFrag fa;
      const unsigned short* pa = aRow + k0 + kk * 32 + hsel * 8;
      fa.q[0] = *(const u32x4*)(pa);
      fa.q[1] = *(const u32x4*)(pa + 16);
#pragma unroll
      for (int t = 0; t < 4; ++t) {
        BfFrag fb;
        const unsigned short* pb = &Bpan[(t * 16 + r) * KPAN + kk * 32 + hsel * 8];
        fb.q[0] = *(const u32x4*)(pb);     // ds_load_b128
        fb.q[1] = *(const u32x4*)(pb + 16);
        acc[t] = __builtin_amdgcn_wmma_f32_16x16x32_bf16(
            false, fa.v, false, fb.v, (short)0, acc[t], false, false);
      }
    }
    __syncthreads();
  }

#pragma unroll
  for (int t = 0; t < 4; ++t) {
#pragma unroll
    for (int rr = 0; rr < 8; ++rr) {
      const int m = m0 + rr + hsel * 8;
      const int j = j0 + t * 16 + r;
      const float v = acc[t][rr];
      if (mode == 1) {
        Gout[(size_t)m * BC + j] = v;
        GoutT[(size_t)j * N_NODES + m] = f2bf(v);
      } else {
        const int bb = j >> 5, c = j & 31;
        Gout[(size_t)m * BC + j] =
            2.0f * v - xin[((size_t)bb * N_NODES + m) * C_IN + c];
      }
    }
  }
}

// ---------------- stage 5: per-node [16x96]@[96x64] + bias (f32 WMMA) -----
__global__ void k_out(const float* __restrict__ xin,
                      const float* __restrict__ G1,
                      const float* __restrict__ G2,
                      const float* __restrict__ W,
                      const float* __restrict__ bias,
                      float* __restrict__ out) {
  const int lane = threadIdx.x & 31;
  const int wave = threadIdx.x >> 5;
  const int hsel = lane >> 4;
  const int r    = lane & 15;
  const int n    = blockIdx.x * 8 + wave;

  const float* xrow  = xin + ((size_t)r * N_NODES + n) * C_IN;  // batch = r
  const float* g1row = G1 + (size_t)n * BC + r * C_IN;
  const float* g2row = G2 + (size_t)n * BC + r * C_IN;
  const float* Wn    = W + (size_t)n * (KW * C_OUT);

  f32x8 acc[4];
#pragma unroll
  for (int t = 0; t < 4; ++t)
    acc[t] = (f32x8){0.f, 0.f, 0.f, 0.f, 0.f, 0.f, 0.f, 0.f};

  for (int kk0 = 0; kk0 < KW; kk0 += 4) {
    const int reg = kk0 >> 5;       // 0: x, 1: G1, 2: G2 (4-blocks never cross)
    const float* abase = (reg == 0) ? xrow : ((reg == 1) ? g1row : g2row);
    const int aoff = (kk0 & 31) + hsel * 2;
    f32x2 a;
    a[0] = abase[aoff];             // A 16x4: v0 = K{0|2}, v1 = K{1|3}
    a[1] = abase[aoff + 1];
    const int kB = kk0 + hsel * 2;
    const float* wb0 = Wn + (size_t)kB * C_OUT;
    const float* wb1 = wb0 + C_OUT;
#pragma unroll
    for (int t = 0; t < 4; ++t) {
      f32x2 bfr;
      bfr[0] = wb0[t * 16 + r];     // B 4x16: v0 = row K{0|2}, v1 = row K{1|3}
      bfr[1] = wb1[t * 16 + r];
      acc[t] = __builtin_amdgcn_wmma_f32_16x16x4_f32(false, a, false, bfr,
                                                     (short)0, acc[t],
                                                     false, false);
    }
  }
#pragma unroll
  for (int t = 0; t < 4; ++t) {
    const int o = t * 16 + r;
    const float bb = bias[o];
#pragma unroll
    for (int rr = 0; rr < 8; ++rr) {
      const int b = rr + hsel * 8;  // C/D: lanes16-31 -> M+8
      out[((size_t)b * N_NODES + n) * C_OUT + o] = acc[t][rr] + bb;
    }
  }
}

// ---------------- host launcher ----------------
extern "C" void kernel_launch(void* const* d_in, const int* in_sizes, int n_in,
                              void* d_out, int out_size, void* d_ws, size_t ws_size,
                              hipStream_t stream) {
  const float* A    = (const float*)d_in[0];
  const float* x    = (const float*)d_in[1];
  const float* cE   = (const float*)d_in[2];
  const float* W    = (const float*)d_in[3];
  const float* bias = (const float*)d_in[4];
  float* out = (float*)d_out;
  const float* A1 = A + (size_t)N_NODES * N_NODES;

  // workspace carve-up (~29.6 MB)
  unsigned short* S   = (unsigned short*)d_ws;                 // [N,N] bf16
  unsigned short* A1h = S   + (size_t)N_NODES * N_NODES;       // [N,N] bf16
  unsigned short* ET  = A1h + (size_t)N_NODES * N_NODES;       // [E,N] bf16
  unsigned short* XT  = ET  + (size_t)E_DIM * N_NODES;         // [BC,N] bf16
  unsigned short* G1T = XT  + (size_t)BC * N_NODES;            // [BC,N] bf16
  float* CEL = (float*)(G1T + (size_t)BC * N_NODES);           // [N,E] f32
  float* G1  = CEL + (size_t)N_NODES * E_DIM;                  // [N,BC] f32
  float* G2  = G1  + (size_t)N_NODES * BC;                     // [N,BC] f32

  k_cvt_bf16<<<(N_NODES * N_NODES + 255) / 256, 256, 0, stream>>>(
      A1, A1h, N_NODES * N_NODES);
  k_build_ET<<<(E_DIM * N_NODES + 255) / 256, 256, 0, stream>>>(cE, ET);
  k_build_XT<<<(BC * N_NODES + 255) / 256, 256, 0, stream>>>(x, XT);

  k_gemm_cel<<<N_NODES / 128, 256, 0, stream>>>(A1h, ET, CEL);
  k_softmax<<<N_NODES, 256, 0, stream>>>(CEL, cE, S);

  k_gemm_sx<<<dim3(BC / 64, N_NODES / 128), 256, 0, stream>>>(
      S, XT, G1, G1T, nullptr, 1);
  k_gemm_sx<<<dim3(BC / 64, N_NODES / 128), 256, 0, stream>>>(
      S, G1T, G2, nullptr, x, 2);

  k_out<<<N_NODES / 8, 256, 0, stream>>>(x, G1, G2, W, bias, out);
}
